// MultiHeadAttention_18915035971585
// MI455X (gfx1250) — compile-verified, matrix operand order checked by reading, not run on hardware
//
#include <hip/hip_runtime.h>
#include <hip/hip_bf16.h>

// ---------------------------------------------------------------------------
// MI455X (gfx1250) multi-head attention, f16 WMMA + f32 accumulate.
// Compute-bound (~103 GFLOP vs ~80MB HBM): all GEMMs via v_wmma_f32_16x16x32_f16.
// Round 3: launch_bounds to kill scratch spills; scale folded into Q proj;
// K fragments consumed tile-at-a-time; float4 LDS strip reads.
// ---------------------------------------------------------------------------

typedef __attribute__((ext_vector_type(16))) _Float16 v16h;
typedef __attribute__((ext_vector_type(8)))  _Float16 h8;
typedef __attribute__((ext_vector_type(8)))  float    v8f;
typedef __attribute__((ext_vector_type(4)))  float    f4;

#define S_LEN  4096
#define DMODEL 1024
#define NHEAD  16
#define DK     64

union Frag16 {
    v16h     v;
    h8       h[2];
    _Float16 e[16];
};

__device__ __forceinline__ v8f wmma_f16(v16h a, v16h b, v8f c) {
    return __builtin_amdgcn_wmma_f32_16x16x32_f16(
        false, a, false, b, (short)0, c, false, false);
}

// A/B fragment from f16 memory: 8 consecutive halves at p and 8 at p+16
// (ISA 16-bit 16x32 matrix layout; baseK folded into p by caller).
__device__ __forceinline__ v16h frag_from_f16(const _Float16* p) {
    Frag16 f;
    f.h[0] = *(const h8*)(p);
    f.h[1] = *(const h8*)(p + 16);
    return f.v;
}

// A fragment from f32 memory with inline f32->f16 conversion.
__device__ __forceinline__ v16h frag_from_f32(const float* p) {
    Frag16 f;
#pragma unroll
    for (int i = 0; i < 8; ++i) f.e[i]     = (_Float16)p[i];
#pragma unroll
    for (int i = 0; i < 8; ++i) f.e[i + 8] = (_Float16)p[i + 16];
    return f.v;
}

__device__ __forceinline__ f4 vmax4(f4 a, f4 b) {
    f4 r;
    r.x = fmaxf(a.x, b.x); r.y = fmaxf(a.y, b.y);
    r.z = fmaxf(a.z, b.z); r.w = fmaxf(a.w, b.w);
    return r;
}

// ---------------------------------------------------------------------------
// Weight conversion f32 -> f16 (row-major NxK == B^T layout for Y = X @ W^T).
// ---------------------------------------------------------------------------
__global__ void cvt_f32_f16(const float* __restrict__ src,
                            _Float16* __restrict__ dst, int n) {
    int i = blockIdx.x * blockDim.x + threadIdx.x;
    if (i < n) dst[i] = (_Float16)src[i];
}

// ---------------------------------------------------------------------------
// GEMM: C[m,n] = (sum_k A[m,k] * W[n,k] + bias[n]) * scaleOut
//   A: f32 [S_LEN x DMODEL]   W: f16 [DMODEL x DMODEL] (N-major)
// One wave per 16x128 tile: 8 WMMAs amortize each A-fragment load.
// MODE 0: f16 out, per-head [h][s][d]     (Q with scale=1/8, K with scale=1)
// MODE 1: f16 out, per-head [h][d][s]     (V^T for the PV WMMA B-fragments)
// MODE 2: f32 out, row-major [s][DMODEL]  (final output)
// ---------------------------------------------------------------------------
template <int MODE>
__global__ void __launch_bounds__(32, 1)
proj_gemm(const float* __restrict__ A,
          const _Float16* __restrict__ W,
          const float* __restrict__ bias,
          void* __restrict__ out, const float scaleOut) {
    const int lane  = threadIdx.x & 31;
    const int m0    = blockIdx.x * 16;
    const int n0    = blockIdx.y * 128;
    const int mr    = lane & 15;
    const int baseK = (lane >> 4) << 3;
    const int m     = m0 + mr;

    v8f acc[8] = {};

    for (int k0 = 0; k0 < DMODEL; k0 += 32) {
        const float* ap = A + (size_t)m * DMODEL + k0 + baseK;
        v16h a = frag_from_f32(ap);
        if (k0 + 32 < DMODEL) __builtin_prefetch(ap + 32, 0, 0);
#pragma unroll
        for (int j = 0; j < 8; ++j) {
            const int n = n0 + j * 16 + mr;
            v16h b = frag_from_f16(W + (size_t)n * DMODEL + k0 + baseK);
            acc[j] = wmma_f16(a, b, acc[j]);
        }
    }

    const int rowBase = m0 + ((lane >> 4) << 3);
#pragma unroll
    for (int j = 0; j < 8; ++j) {
        const int n  = n0 + j * 16 + mr;
        const float bv = bias[n];
#pragma unroll
        for (int r = 0; r < 8; ++r) {
            const int row = rowBase + r;
            float val = (acc[j][r] + bv) * scaleOut;
            if (MODE == 2) {
                ((float*)out)[(size_t)row * DMODEL + n] = val;
            } else {
                const int h = n >> 6;
                const int d = n & 63;
                _Float16 hv = (_Float16)val;
                if (MODE == 0)
                    ((_Float16*)out)[((size_t)h * S_LEN + row) * DK + d] = hv;
                else
                    ((_Float16*)out)[((size_t)h * DK + d) * S_LEN + row] = hv;
            }
        }
    }
}

// ---------------------------------------------------------------------------
// Flash attention: one wave per (32-query block, head). Two 16-row M-tiles
// share every K/V fragment. Softmax is a row-per-lane LDS pass: lane l owns
// score row l; running max/denom are lane scalars (no cross-lane shuffles).
// 1/sqrt(dk) is pre-folded into Q, so score D-tiles go straight to LDS.
// ---------------------------------------------------------------------------
__global__ void __launch_bounds__(32, 1)
attention_kernel(const _Float16* __restrict__ Qh,
                 const _Float16* __restrict__ Kh,
                 const _Float16* __restrict__ Vt,
                 float* __restrict__ Obuf) {
    __shared__ __align__(16) float    Slds[2][16][32];   // raw scores
    __shared__ __align__(16) _Float16 Plds[2][16][32];   // exp(s - m) as f16
    __shared__ __align__(16) float    rowLds[32];        // alpha / denom strip

    const int lane  = threadIdx.x & 31;
    const int q0    = blockIdx.x * 32;
    const int h     = blockIdx.y;
    const int mr    = lane & 15;
    const int hi    = lane >> 4;
    const int baseK = hi << 3;

    const _Float16* Qbase = Qh + (size_t)h * S_LEN * DK;
    const _Float16* Kbase = Kh + (size_t)h * S_LEN * DK;
    const _Float16* Vbase = Vt + (size_t)h * DK * S_LEN;

    // Q A-fragments: 2 M-tiles x (K=64 -> two K=32 fragments)
    v16h qa[2][2];
#pragma unroll
    for (int qt = 0; qt < 2; ++qt) {
        const _Float16* qp = Qbase + (size_t)(q0 + qt * 16 + mr) * DK + baseK;
        qa[qt][0] = frag_from_f16(qp);
        qa[qt][1] = frag_from_f16(qp + 32);
    }

    v8f acc[2][4] = {};
    float m_run = -1e30f;   // per-lane: lane == score row (qt = l>>4, r = l&15)
    float l_run = 0.f;

    for (int kc = 0; kc < S_LEN; kc += 32) {
        // ---- scores: per key-tile, load K fragments then consume at once ----
#pragma unroll
        for (int t = 0; t < 2; ++t) {
            const _Float16* kp = Kbase + (size_t)(kc + t * 16 + mr) * DK + baseK;
            v16h kb0 = frag_from_f16(kp);
            v16h kb1 = frag_from_f16(kp + 32);
#pragma unroll
            for (int qt = 0; qt < 2; ++qt) {
                v8f z = {};
                z = wmma_f16(qa[qt][0], kb0, z);
                z = wmma_f16(qa[qt][1], kb1, z);
#pragma unroll
                for (int r = 0; r < 8; ++r)
                    Slds[qt][hi * 8 + r][t * 16 + mr] = z[r];
            }
        }
        if (kc + 32 < S_LEN) {  // overlap next K rows with the softmax pass
            __builtin_prefetch(Kbase + (size_t)(kc + 32 + mr) * DK, 0, 0);
            __builtin_prefetch(Kbase + (size_t)(kc + 48 + mr) * DK, 0, 0);
        }
        __syncthreads();

        // ---- row-per-lane online softmax ----
        {
            const f4* srow = (const f4*)&Slds[lane >> 4][lane & 15][0];
            f4 x[8];
#pragma unroll
            for (int i = 0; i < 8; ++i) x[i] = srow[i];
            f4 mm = vmax4(vmax4(vmax4(x[0], x[1]), vmax4(x[2], x[3])),
                          vmax4(vmax4(x[4], x[5]), vmax4(x[6], x[7])));
            const float tm = fmaxf(fmaxf(mm.x, mm.y), fmaxf(mm.z, mm.w));
            const float mn = fmaxf(m_run, tm);
            const float alpha = __expf(m_run - mn);
            m_run = mn;

            Frag16 pk[2];
            f4 sum = {0.f, 0.f, 0.f, 0.f};
#pragma unroll
            for (int i = 0; i < 8; ++i) {
                f4 p;
                p.x = __expf(x[i].x - mn);
                p.y = __expf(x[i].y - mn);
                p.z = __expf(x[i].z - mn);
                p.w = __expf(x[i].w - mn);
                sum.x += p.x; sum.y += p.y; sum.z += p.z; sum.w += p.w;
                const int half = i >> 2, e0 = (i & 3) * 4;
                pk[half].e[e0 + 0] = (_Float16)p.x;
                pk[half].e[e0 + 1] = (_Float16)p.y;
                pk[half].e[e0 + 2] = (_Float16)p.z;
                pk[half].e[e0 + 3] = (_Float16)p.w;
            }
            l_run = l_run * alpha + (sum.x + sum.y) + (sum.z + sum.w);

            h8* prow = (h8*)&Plds[lane >> 4][lane & 15][0];
            prow[0] = pk[0].h[0];
            prow[1] = pk[0].h[1];
            prow[2] = pk[1].h[0];
            prow[3] = pk[1].h[1];
            rowLds[lane] = alpha;
        }
        __syncthreads();

        // ---- rescale accumulators (alpha per row, float4 LDS reads) ----
#pragma unroll
        for (int qt = 0; qt < 2; ++qt) {
            const f4* av4 = (const f4*)&rowLds[qt * 16 + hi * 8];
            f4 a0 = av4[0], a1 = av4[1];
            float av[8] = {a0.x, a0.y, a0.z, a0.w, a1.x, a1.y, a1.z, a1.w};
#pragma unroll
            for (int j = 0; j < 4; ++j)
#pragma unroll
                for (int r = 0; r < 8; ++r) acc[qt][j][r] *= av[r];
        }

        // ---- O += P @ V : V fragment shared by both M-tiles ----
        v16h pa0 = frag_from_f16(&Plds[0][mr][baseK]);
        v16h pa1 = frag_from_f16(&Plds[1][mr][baseK]);
#pragma unroll
        for (int j = 0; j < 4; ++j) {
            const int d = j * 16 + mr;
            v16h vb = frag_from_f16(Vbase + (size_t)d * S_LEN + kc + baseK);
            acc[0][j] = wmma_f16(pa0, vb, acc[0][j]);
            acc[1][j] = wmma_f16(pa1, vb, acc[1][j]);
        }
        __syncthreads();  // protect Slds/Plds/rowLds before next iteration
    }

    // ---- normalize: denominators through the LDS strip, then store ----
    rowLds[lane] = l_run;
    __syncthreads();
#pragma unroll
    for (int qt = 0; qt < 2; ++qt) {
        const f4* lv4 = (const f4*)&rowLds[qt * 16 + hi * 8];
        f4 l0 = lv4[0], l1 = lv4[1];
        float linv[8] = {1.f / l0.x, 1.f / l0.y, 1.f / l0.z, 1.f / l0.w,
                         1.f / l1.x, 1.f / l1.y, 1.f / l1.z, 1.f / l1.w};
#pragma unroll
        for (int j = 0; j < 4; ++j)
#pragma unroll
            for (int r = 0; r < 8; ++r) {
                const int row = q0 + qt * 16 + hi * 8 + r;
                const int d   = j * 16 + mr;
                Obuf[(size_t)row * DMODEL + h * DK + d] = acc[qt][j][r] * linv[r];
            }
    }
}

// ---------------------------------------------------------------------------
extern "C" void kernel_launch(void* const* d_in, const int* in_sizes, int n_in,
                              void* d_out, int out_size, void* d_ws, size_t ws_size,
                              hipStream_t stream) {
    (void)in_sizes; (void)n_in; (void)out_size; (void)ws_size;

    const float* query = (const float*)d_in[0];
    const float* key   = (const float*)d_in[1];
    const float* value = (const float*)d_in[2];
    const float* W_q   = (const float*)d_in[3];
    const float* b_q   = (const float*)d_in[4];
    const float* W_k   = (const float*)d_in[5];
    const float* b_k   = (const float*)d_in[6];
    const float* W_v   = (const float*)d_in[7];
    const float* b_v   = (const float*)d_in[8];
    const float* W_o   = (const float*)d_in[9];
    const float* b_o   = (const float*)d_in[10];
    float* out = (float*)d_out;

    char* ws = (char*)d_ws;
    const size_t MB = 1u << 20;
    _Float16* Wq16 = (_Float16*)(ws + 0 * MB);   // 2MB each
    _Float16* Wk16 = (_Float16*)(ws + 2 * MB);
    _Float16* Wv16 = (_Float16*)(ws + 4 * MB);
    _Float16* Wo16 = (_Float16*)(ws + 6 * MB);
    _Float16* Qh   = (_Float16*)(ws + 8 * MB);   // 8MB  [h][s][d], pre-scaled 1/8
    _Float16* Kh   = (_Float16*)(ws + 16 * MB);  // 8MB  [h][s][d]
    _Float16* Vt   = (_Float16*)(ws + 24 * MB);  // 8MB  [h][d][s]
    float*    attn = (float*)(ws + 32 * MB);     // 16MB [s][DMODEL]

    const int NW = DMODEL * DMODEL;
    cvt_f32_f16<<<NW / 256, 256, 0, stream>>>(W_q, Wq16, NW);
    cvt_f32_f16<<<NW / 256, 256, 0, stream>>>(W_k, Wk16, NW);
    cvt_f32_f16<<<NW / 256, 256, 0, stream>>>(W_v, Wv16, NW);
    cvt_f32_f16<<<NW / 256, 256, 0, stream>>>(W_o, Wo16, NW);

    dim3 gProj(S_LEN / 16, DMODEL / 128);  // 256 x 8 tiles of 16x128
    proj_gemm<0><<<gProj, 32, 0, stream>>>(query, Wq16, b_q, (void*)Qh, 0.125f);
    proj_gemm<0><<<gProj, 32, 0, stream>>>(key,   Wk16, b_k, (void*)Kh, 1.0f);
    proj_gemm<1><<<gProj, 32, 0, stream>>>(value, Wv16, b_v, (void*)Vt, 1.0f);

    dim3 gAttn(S_LEN / 32, NHEAD);  // 128 query blocks x 16 heads
    attention_kernel<<<gAttn, 32, 0, stream>>>(Qh, Kh, Vt, attn);

    proj_gemm<2><<<gProj, 32, 0, stream>>>(attn, Wo16, b_o, (void*)out, 1.0f);
}
